// SimpleTwoLevelGNN_89232240542334
// MI455X (gfx1250) — matrix operand order repaired and checked by reading, not compiled, not run on hardware
//
#include <hip/hip_runtime.h>
#include <hip/hip_bf16.h>
#include <stdint.h>

// ---------------- problem constants (match reference) ----------------
#define N_NODES_C 100000
#define N_EDGES_C 200000
#define BGRAPH    2048
#define ND        64
#define ED        16
#define HID       512
#define OUTD      512
#define HEADS     4
#define BN_SCALE  0.9999950000374997f   // 1/sqrt(1 + 1e-5)

typedef __bf16 bf16x8  __attribute__((ext_vector_type(8)));
typedef __bf16 bf16x16 __attribute__((ext_vector_type(16)));
typedef float  f32x8   __attribute__((ext_vector_type(8)));
typedef unsigned int u32x4 __attribute__((ext_vector_type(4)));
typedef int          i32x4 __attribute__((ext_vector_type(4)));
typedef int          i32x8 __attribute__((ext_vector_type(8)));

__device__ __forceinline__ __bf16 f2bf(float f) {
  union { float f; uint32_t u; } v; v.f = f;
  uint32_t r = v.u + 0x7FFFu + ((v.u >> 16) & 1u);   // round-to-nearest-even
  union { uint16_t u; __bf16 b; } o; o.u = (uint16_t)(r >> 16);
  return o.b;
}

// =====================================================================
// TDM: DMA a [tile_rows x 32] bf16 tile (row-major, row stride = K elems)
// from global into LDS at lds_addr. tensor_dim0/1 are the REMAINING
// cols/rows from the tile origin so hardware OOB returns zero-fill for
// the ragged M edge. Issued by one wave; tracked by TENSORcnt.
// D# bit layout per CDNA5 ISA ch.8 (group0 128b, group1 256b, 2D mode).
// =====================================================================
__device__ __forceinline__ void tdm_load_tile_bf16(
    uint32_t lds_addr, const __bf16* tile_origin,
    uint32_t cols_rem, uint32_t rows_rem, uint32_t row_stride_elems,
    uint32_t tile_rows)
{
  uint64_t ga = (uint64_t)(uintptr_t)tile_origin;
  u32x4 g0;
  g0[0] = 1u;                                           // count=1, user mode
  g0[1] = lds_addr;                                     // LDS byte address
  g0[2] = (uint32_t)ga;                                 // global_addr[31:0]
  g0[3] = (uint32_t)((ga >> 32) & 0x01FFFFFFu)          // global_addr[56:32]
        | (2u << 30);                                   // type = 2 ("image")
  i32x8 g1;
  g1[0] = (int)(1u << 16);                              // data_size=1 (2 bytes)
  g1[1] = (int)((cols_rem & 0xFFFFu) << 16);            // tensor_dim0[15:0]
  g1[2] = (int)((cols_rem >> 16) | ((rows_rem & 0xFFFFu) << 16)); // td0 hi, td1 lo
  g1[3] = (int)((rows_rem >> 16) | (32u << 16));        // td1 hi | tile_dim0=32
  g1[4] = (int)(tile_rows & 0xFFFFu);                   // tile_dim1 (tile_dim2=0)
  g1[5] = (int)row_stride_elems;                        // tensor_dim0_stride lo
  g1[6] = 0;                                            // stride hi | td1_stride lo
  g1[7] = 0;
  i32x4 z4 = {0, 0, 0, 0};                              // group 2 (unused, 2D)
  i32x8 z8 = {0, 0, 0, 0, 0, 0, 0, 0};                  // group 3 (unused, 2D)
  __builtin_amdgcn_tensor_load_to_lds(g0, g1, z4, z4, z8, 0);
}

// =====================================================================
// bf16 WMMA GEMM:  C[M,N] = epilogue( A @ W + bias )
// A: [M,K] bf16 (pre-converted, any elementwise add already fused),
// W: [K,N] bf16. Block = 128 threads (4 waves). Tile 128x64, K-step 32.
// A tiles arrive via double-buffered TENSOR_LOAD_TO_LDS (wave 0 issues,
// s_wait_tensorcnt paces); W tile staged transposed by all threads.
// Each wave owns a 32-row strip: 8 x v_wmma_f32_16x16x32_bf16 per K-step.
// =====================================================================
__global__ __launch_bounds__(128)
void wmma_gemm_kernel(const __bf16* __restrict__ A, const __bf16* __restrict__ W,
                      const float* __restrict__ bias,
                      const float* __restrict__ bn_g, const float* __restrict__ bn_b,
                      float* __restrict__ C, int M, int N, int K, int fuse_bn_relu)
{
  __shared__ __align__(16) __bf16 As[2][128][32];   // [buf][m][k] (TDM dest)
  __shared__ __align__(16) __bf16 Bs[64][32];       // [n][k] transposed stage

  const int tid    = threadIdx.x;
  const int m_base = blockIdx.x * 128;
  const int n_base = blockIdx.y * 64;
  const int wave   = tid >> 5;          // 0..3 (wave32)
  const int lane   = tid & 31;
  const int lhalf  = lane >> 4;
  const int l16    = lane & 15;
  const bool wave0 = (tid < 32);

  // LDS byte addresses for the TDM descriptor: flat-aperture addr[31:0]
  // is the LDS offset (ISA 10.2 aperture mapping).
  const uint32_t ldsA0 = (uint32_t)(uintptr_t)&As[0][0][0];
  const uint32_t ldsA1 = (uint32_t)(uintptr_t)&As[1][0][0];

  const uint32_t rows_rem = (uint32_t)(M - m_base);   // grid guarantees > 0
  const __bf16* Abase = A + (size_t)m_base * K;

  f32x8 acc[2][4] = {};
  const int nk = K >> 5;

  if (wave0)   // prefetch first A tile
    tdm_load_tile_bf16(ldsA0, Abase, (uint32_t)K, rows_rem, (uint32_t)K, 128);

  int buf = 0;
  for (int kt = 0; kt < nk; ++kt) {
    const int k0 = kt << 5;
    // ---- stage W tile (32x64) transposed into Bs[n][k] (bf16 moves) ----
    {
      int k_l  = tid & 31;
      int nseg = (tid >> 5) << 4;
      const __bf16* wp = W + (size_t)(k0 + k_l) * N + n_base + nseg;
      bf16x8 w0 = *reinterpret_cast<const bf16x8*>(wp);
      bf16x8 w1 = *reinterpret_cast<const bf16x8*>(wp + 8);
      #pragma unroll
      for (int i = 0; i < 8; ++i) {
        Bs[nseg + i][k_l]     = w0[i];
        Bs[nseg + 8 + i][k_l] = w1[i];
      }
    }
    // ---- overlap: DMA next A tile into the other buffer ----
    if (wave0) {
      if (kt + 1 < nk) {
        tdm_load_tile_bf16(buf ? ldsA0 : ldsA1, Abase + (k0 + 32),
                           (uint32_t)(K - k0 - 32), rows_rem, (uint32_t)K, 128);
        __builtin_amdgcn_s_wait_tensorcnt(1);   // current tile done, prefetch in flight
      } else {
        __builtin_amdgcn_s_wait_tensorcnt(0);   // final tile: drain everything
      }
    }
    __syncthreads();

    // ---- compute: 2 A-fragments x 4 B-fragments = 8 WMMA per wave ----
    #pragma unroll
    for (int f = 0; f < 2; ++f) {
      // A frag: lane L row = 32*wave + 16*f + L%16; elem 0..7 = K{0..7|8..15},
      // elem 8..15 = K{16..23|24..31} selected by lane half (ISA 7.12.2).
      int arow = (wave << 5) + (f << 4) + l16;
      bf16x8 alo = *reinterpret_cast<const bf16x8*>(&As[buf][arow][lhalf << 3]);
      bf16x8 ahi = *reinterpret_cast<const bf16x8*>(&As[buf][arow][16 + (lhalf << 3)]);
      bf16x16 afrag = __builtin_shufflevector(alo, ahi,
          0,1,2,3,4,5,6,7,8,9,10,11,12,13,14,15);
      #pragma unroll
      for (int t = 0; t < 4; ++t) {
        int bcol = (t << 4) + l16;
        bf16x8 blo = *reinterpret_cast<const bf16x8*>(&Bs[bcol][lhalf << 3]);
        bf16x8 bhi = *reinterpret_cast<const bf16x8*>(&Bs[bcol][16 + (lhalf << 3)]);
        bf16x16 bfrag = __builtin_shufflevector(blo, bhi,
            0,1,2,3,4,5,6,7,8,9,10,11,12,13,14,15);
        acc[f][t] = __builtin_amdgcn_wmma_f32_16x16x32_bf16(
            false, afrag, false, bfrag, (short)0, acc[f][t], false, false);
      }
    }
    __syncthreads();
    buf ^= 1;
  }

  // ---- epilogue: C/D layout lane L vgpr i -> row = i + 8*lhalf, col = L%16
  #pragma unroll
  for (int f = 0; f < 2; ++f) {
    #pragma unroll
    for (int t = 0; t < 4; ++t) {
      int col = n_base + (t << 4) + l16;
      float bc = bias ? bias[col] : 0.0f;
      float g  = fuse_bn_relu ? bn_g[col] : 0.0f;
      float bb = fuse_bn_relu ? bn_b[col] : 0.0f;
      #pragma unroll
      for (int i = 0; i < 8; ++i) {
        int row = m_base + (wave << 5) + (f << 4) + (lhalf << 3) + i;
        if (row < M) {
          float v = acc[f][t][i] + bc;
          if (fuse_bn_relu) v = fmaxf(g * (v * BN_SCALE) + bb, 0.0f);
          C[(size_t)row * N + col] = v;
        }
      }
    }
  }
}

// =====================================================================
// Streaming f32 -> bf16 conversion with optional elementwise add
// (fuses the GINE "x + aggr" residual). 8 elements per thread.
// =====================================================================
__global__ __launch_bounds__(256)
void cvt_bf16_kernel(const float* __restrict__ a, const float* __restrict__ a2,
                     __bf16* __restrict__ out, long long n)
{
  long long base = ((long long)blockIdx.x * blockDim.x + threadIdx.x) * 8;
  if (base >= n) return;
  float v[8];
  *reinterpret_cast<float4*>(&v[0]) = *reinterpret_cast<const float4*>(a + base);
  *reinterpret_cast<float4*>(&v[4]) = *reinterpret_cast<const float4*>(a + base + 4);
  if (a2) {
    float4 t0 = *reinterpret_cast<const float4*>(a2 + base);
    float4 t1 = *reinterpret_cast<const float4*>(a2 + base + 4);
    v[0] += t0.x; v[1] += t0.y; v[2] += t0.z; v[3] += t0.w;
    v[4] += t1.x; v[5] += t1.y; v[6] += t1.z; v[7] += t1.w;
  }
  bf16x8 o;
  #pragma unroll
  for (int i = 0; i < 8; ++i) o[i] = f2bf(v[i]);
  *reinterpret_cast<bf16x8*>(out + base) = o;
}

// =====================================================================
// GINE1 edge messages: m = relu(x[src] + edge_attr @ e1_W + e1_b),
// atomic scatter-add into aggr[dst]. 4 edges per 256-thread block.
// =====================================================================
__global__ __launch_bounds__(256)
void gine1_edge_kernel(const float* __restrict__ x, const int* __restrict__ eidx,
                       const float* __restrict__ eattr,
                       const float* __restrict__ eW, const float* __restrict__ eb,
                       float* __restrict__ aggr)
{
  int tid = threadIdx.x;
  int e = blockIdx.x * 4 + (tid >> 6);
  int c = tid & 63;
  if (e >= N_EDGES_C) return;
  int src = eidx[e];
  int dst = eidx[N_EDGES_C + e];
  const float* ea = eattr + (size_t)e * ED;
  float acc = eb[c];
  #pragma unroll
  for (int k = 0; k < ED; ++k) acc += ea[k] * eW[k * ND + c];
  float m = fmaxf(x[(size_t)src * ND + c] + acc, 0.0f);
  atomicAdd(&aggr[(size_t)dst * ND + c], m);
}

// GINE2 edge messages: one edge per block, 512 channels, 2 per thread.
__global__ __launch_bounds__(256)
void gine2_edge_kernel(const float* __restrict__ h, const int* __restrict__ eidx,
                       const float* __restrict__ eattr,
                       const float* __restrict__ eW, const float* __restrict__ eb,
                       float* __restrict__ aggr)
{
  __shared__ float ea[ED];
  int e = blockIdx.x;
  int tid = threadIdx.x;
  if (tid < ED) ea[tid] = eattr[(size_t)e * ED + tid];
  __syncthreads();
  int src = eidx[e];
  int dst = eidx[N_EDGES_C + e];
  #pragma unroll
  for (int r = 0; r < 2; ++r) {
    int c = tid + r * 256;
    float acc = eb[c];
    #pragma unroll
    for (int k = 0; k < ED; ++k) acc += ea[k] * eW[k * HID + c];
    float m = fmaxf(h[(size_t)src * HID + c] + acc, 0.0f);
    atomicAdd(&aggr[(size_t)dst * HID + c], m);
  }
}

// =====================================================================
// Global mean pool helpers
// =====================================================================
__global__ void pool_accum_kernel(const float* __restrict__ h2,
                                  const int* __restrict__ batch,
                                  float* __restrict__ sums)
{
  int idx = blockIdx.x * blockDim.x + threadIdx.x;   // node*128 + chunk4
  if (idx >= N_NODES_C * 128) return;
  int node  = idx >> 7;
  int chunk = idx & 127;
  int b = batch[node];
  const float* hp = h2 + (size_t)node * OUTD + chunk * 4;
  float* sp = sums + (size_t)b * OUTD + chunk * 4;
  float4 v = *reinterpret_cast<const float4*>(hp);
  atomicAdd(sp + 0, v.x); atomicAdd(sp + 1, v.y);
  atomicAdd(sp + 2, v.z); atomicAdd(sp + 3, v.w);
}

__global__ void pool_count_kernel(const int* __restrict__ batch, float* __restrict__ cnts)
{
  int i = blockIdx.x * blockDim.x + threadIdx.x;
  if (i < N_NODES_C) atomicAdd(&cnts[batch[i]], 1.0f);
}

// Build xn[b][0..6][c] = {pooled, ecfp, topo, maccs, estate, rdkit2d, phar2d}
__global__ void stack_xn_kernel(const float* __restrict__ sums, const float* __restrict__ cnts,
                                const float* __restrict__ f1, const float* __restrict__ f2,
                                const float* __restrict__ f3, const float* __restrict__ f4,
                                const float* __restrict__ f5, const float* __restrict__ f6,
                                float* __restrict__ xn)
{
  int idx = blockIdx.x * blockDim.x + threadIdx.x;
  if (idx >= BGRAPH * 7 * OUTD) return;
  int c = idx % OUTD;
  int n = (idx / OUTD) % 7;
  int b = idx / (7 * OUTD);
  size_t o = (size_t)b * OUTD + c;
  float v;
  switch (n) {
    case 0:  v = sums[o] / fmaxf(cnts[b], 1.0f); break;
    case 1:  v = f1[o]; break;
    case 2:  v = f2[o]; break;
    case 3:  v = f3[o]; break;
    case 4:  v = f4[o]; break;
    case 5:  v = f5[o]; break;
    default: v = f6[o]; break;
  }
  xn[idx] = v;
}

// =====================================================================
// GATv2 attention over the fixed 13-edge / 7-node dummy graph.
// gl, gr: [B*7, HEADS*HID] (row = b*7+n, col = h*HID+c).
// Topology: edges 0..5: 0->n+1 ; edge 6: 0->0 ; edges 7..12: self loops.
// =====================================================================
__device__ __forceinline__ int edge_src(int e) { return (e < 7) ? 0 : (e - 6); }
__device__ __forceinline__ int edge_dst(int e) { return (e < 6) ? (e + 1) : ((e == 6) ? 0 : (e - 6)); }

__global__ __launch_bounds__(256)
void gatv2_att_kernel(const float* __restrict__ gl, const float* __restrict__ gr,
                      const float* __restrict__ att, const float* __restrict__ bias,
                      const float* __restrict__ bn_g, const float* __restrict__ bn_b,
                      float* __restrict__ zout)
{
  __shared__ float lg[13][HEADS];
  __shared__ float al[13][HEADS];
  int b = blockIdx.x;
  int tid = threadIdx.x;
  int wave = tid >> 5, lane = tid & 31;
  size_t base = (size_t)b * 7 * (HEADS * HID);

  // logits: 52 (edge, head) dot products of length HID
  for (int p = wave; p < 13 * HEADS; p += 8) {
    int e = p / HEADS, hd = p % HEADS;
    const float* glp = gl + base + (size_t)edge_src(e) * (HEADS * HID) + hd * HID;
    const float* grp = gr + base + (size_t)edge_dst(e) * (HEADS * HID) + hd * HID;
    const float* ap  = att + hd * HID;
    float acc = 0.0f;
    for (int c = lane; c < HID; c += 32) {
      float s = glp[c] + grp[c];
      s = (s > 0.0f) ? s : 0.2f * s;       // leaky_relu(0.2)
      acc += s * ap[c];
    }
    #pragma unroll
    for (int o = 16; o > 0; o >>= 1) acc += __shfl_down(acc, o, 32);
    if (lane == 0) lg[e][hd] = acc;
  }
  __syncthreads();

  // segment softmax per (dst node, head)
  if (tid < 7 * HEADS) {
    int n = tid / HEADS, hd = tid % HEADS;
    if (n == 0) {
      al[6][hd] = 1.0f;                    // single incoming edge (self loop)
    } else {
      int e1 = n - 1, e2 = n + 6;
      float l1 = lg[e1][hd], l2 = lg[e2][hd];
      float mx = fmaxf(l1, l2);
      float x1 = __expf(l1 - mx), x2 = __expf(l2 - mx);
      float s = x1 + x2;
      al[e1][hd] = x1 / s;
      al[e2][hd] = x2 / s;
    }
  }
  __syncthreads();

  // out[b][n][c] = relu(bn( mean_h( sum_{e->n} alpha*gl_src ) + bias ))
  for (int v = tid; v < 7 * OUTD; v += 256) {
    int n = v / OUTD, c = v % OUTD;
    float acc = 0.0f;
    if (n == 0) {
      const float* g0 = gl + base;
      #pragma unroll
      for (int hd = 0; hd < HEADS; ++hd)
        acc += al[6][hd] * g0[hd * HID + c];
    } else {
      const float* g0 = gl + base;
      const float* gn = gl + base + (size_t)n * (HEADS * HID);
      #pragma unroll
      for (int hd = 0; hd < HEADS; ++hd)
        acc += al[n - 1][hd] * g0[hd * HID + c] + al[n + 6][hd] * gn[hd * HID + c];
    }
    acc = acc * (1.0f / HEADS) + bias[c];
    acc = fmaxf(bn_g[c] * (acc * BN_SCALE) + bn_b[c], 0.0f);
    zout[(size_t)b * 7 * OUTD + v] = acc;
  }
}

// Final FC: out[b] = z[b,0,:] @ fc_W + fc_b
__global__ __launch_bounds__(256)
void fc_kernel(const float* __restrict__ z, const float* __restrict__ W,
               const float* __restrict__ bptr, float* __restrict__ out)
{
  __shared__ float red[8];
  int b = blockIdx.x, tid = threadIdx.x;
  const float* zp = z + (size_t)b * 7 * OUTD;    // node 0 row
  float acc = 0.0f;
  for (int c = tid; c < HID; c += 256) acc += zp[c] * W[c];
  #pragma unroll
  for (int o = 16; o > 0; o >>= 1) acc += __shfl_down(acc, o, 32);
  if ((tid & 31) == 0) red[tid >> 5] = acc;
  __syncthreads();
  if (tid == 0) {
    float s = 0.0f;
    #pragma unroll
    for (int i = 0; i < 8; ++i) s += red[i];
    out[b] = s + bptr[0];
  }
}

// =====================================================================
static inline void run_cvt(const float* a, const float* a2, __bf16* out,
                           long long n, hipStream_t s)
{
  long long threads = n / 8;
  cvt_bf16_kernel<<<(unsigned)((threads + 255) / 256), 256, 0, s>>>(a, a2, out, n);
}

extern "C" void kernel_launch(void* const* d_in, const int* in_sizes, int n_in,
                              void* d_out, int out_size, void* d_ws, size_t ws_size,
                              hipStream_t stream)
{
  (void)in_sizes; (void)n_in; (void)out_size; (void)ws_size;

  const float* x       = (const float*)d_in[0];
  const int*   eidx    = (const int*)  d_in[1];
  const float* eattr   = (const float*)d_in[2];
  const int*   batch   = (const int*)  d_in[3];
  const float* ecfp    = (const float*)d_in[4];
  const float* topo    = (const float*)d_in[5];
  const float* maccs   = (const float*)d_in[6];
  const float* estate  = (const float*)d_in[7];
  const float* rdkit2d = (const float*)d_in[8];
  const float* phar2d  = (const float*)d_in[9];
  const float* g1_W = (const float*)d_in[10], *g1_b = (const float*)d_in[11];
  const float* e1_W = (const float*)d_in[12], *e1_b = (const float*)d_in[13];
  const float* bn1_g = (const float*)d_in[14], *bn1_b = (const float*)d_in[15];
  const float* g2_W = (const float*)d_in[16], *g2_b = (const float*)d_in[17];
  const float* e2_W = (const float*)d_in[18], *e2_b = (const float*)d_in[19];
  const float* bn2_g = (const float*)d_in[20], *bn2_b = (const float*)d_in[21];
  const float* gat1_Wl = (const float*)d_in[22], *gat1_bl = (const float*)d_in[23];
  const float* gat1_Wr = (const float*)d_in[24], *gat1_br = (const float*)d_in[25];
  const float* gat1_att = (const float*)d_in[26], *gat1_bias = (const float*)d_in[27];
  const float* bn3_g = (const float*)d_in[28], *bn3_b = (const float*)d_in[29];
  const float* gat2_Wl = (const float*)d_in[30], *gat2_bl = (const float*)d_in[31];
  const float* gat2_Wr = (const float*)d_in[32], *gat2_br = (const float*)d_in[33];
  const float* gat2_att = (const float*)d_in[34], *gat2_bias = (const float*)d_in[35];
  const float* bn4_g = (const float*)d_in[36], *bn4_b = (const float*)d_in[37];
  const float* fc_W = (const float*)d_in[38], *fc_b = (const float*)d_in[39];

  // ---- workspace layout (256B-aligned regions, heavily reused) ----
  char* ws = (char*)d_ws;
  const size_t SZF = (size_t)N_NODES_C * OUTD * sizeof(float);     // 204.8 MB
  float*  R0   = (float*)(ws);                       // h -> gl
  float*  R1   = (float*)(ws + SZF);                 // aggr1/aggr2 -> xn/z1/z2
  float*  R2   = (float*)(ws + 2 * SZF);             // h2 -> gr
  __bf16* Abf  = (__bf16*)(ws + 3 * SZF);            // bf16 activations (102.4 MB)
  __bf16* Wbf  = (__bf16*)(ws + 3 * SZF + SZF / 2);  // bf16 weights (<= 2 MB)
  float*  sums = (float*)(ws + 3 * SZF + SZF / 2 + ((size_t)4 << 20));
  float*  cnts = sums + (size_t)BGRAPH * OUTD;

  const int MG = BGRAPH * 7;                         // 14336 GAT rows

  // ---- GINE layer 1 ----
  float* aggr1 = R1;
  (void)hipMemsetAsync(aggr1, 0, (size_t)N_NODES_C * ND * sizeof(float), stream);
  gine1_edge_kernel<<<(N_EDGES_C + 3) / 4, 256, 0, stream>>>(
      x, eidx, eattr, e1_W, e1_b, aggr1);
  run_cvt(x, aggr1, Abf, (long long)N_NODES_C * ND, stream);       // A = x + aggr
  run_cvt(g1_W, nullptr, Wbf, (long long)ND * HID, stream);
  {
    dim3 g((N_NODES_C + 127) / 128, HID / 64);
    wmma_gemm_kernel<<<g, 128, 0, stream>>>(Abf, Wbf, g1_b, bn1_g, bn1_b,
                                            R0, N_NODES_C, HID, ND, 1);
  }

  // ---- GINE layer 2 ----
  float* aggr2 = R1;
  (void)hipMemsetAsync(aggr2, 0, SZF, stream);
  gine2_edge_kernel<<<N_EDGES_C, 256, 0, stream>>>(
      R0, eidx, eattr, e2_W, e2_b, aggr2);
  run_cvt(R0, aggr2, Abf, (long long)N_NODES_C * HID, stream);     // A = h + aggr
  run_cvt(g2_W, nullptr, Wbf, (long long)HID * OUTD, stream);
  {
    dim3 g((N_NODES_C + 127) / 128, OUTD / 64);
    wmma_gemm_kernel<<<g, 128, 0, stream>>>(Abf, Wbf, g2_b, bn2_g, bn2_b,
                                            R2, N_NODES_C, OUTD, HID, 1);
  }

  // ---- global mean pool + stack dummy graphs ----
  (void)hipMemsetAsync(sums, 0, (size_t)BGRAPH * OUTD * sizeof(float) + BGRAPH * sizeof(float), stream);
  pool_accum_kernel<<<(N_NODES_C * 128 + 255) / 256, 256, 0, stream>>>(R2, batch, sums);
  pool_count_kernel<<<(N_NODES_C + 255) / 256, 256, 0, stream>>>(batch, cnts);
  float* xn = R1;   // aggr2 is dead now
  stack_xn_kernel<<<(BGRAPH * 7 * OUTD + 255) / 256, 256, 0, stream>>>(
      sums, cnts, ecfp, topo, maccs, estate, rdkit2d, phar2d, xn);

  // ---- GATv2 layer 1 ----
  float* gl = R0;                          // h is dead
  float* gr = R2;                          // h2 is dead (pool done)
  run_cvt(xn, nullptr, Abf, (long long)MG * OUTD, stream);
  {
    dim3 g(MG / 128, (HEADS * HID) / 64);
    run_cvt(gat1_Wl, nullptr, Wbf, (long long)OUTD * HEADS * HID, stream);
    wmma_gemm_kernel<<<g, 128, 0, stream>>>(Abf, Wbf, gat1_bl, nullptr, nullptr,
                                            gl, MG, HEADS * HID, OUTD, 0);
    run_cvt(gat1_Wr, nullptr, Wbf, (long long)OUTD * HEADS * HID, stream);
    wmma_gemm_kernel<<<g, 128, 0, stream>>>(Abf, Wbf, gat1_br, nullptr, nullptr,
                                            gr, MG, HEADS * HID, OUTD, 0);
  }
  float* z1 = (float*)(ws + SZF + ((size_t)32 << 20));
  gatv2_att_kernel<<<BGRAPH, 256, 0, stream>>>(gl, gr, gat1_att, gat1_bias,
                                               bn3_g, bn3_b, z1);

  // ---- GATv2 layer 2 ----
  run_cvt(z1, nullptr, Abf, (long long)MG * HID, stream);
  {
    dim3 g(MG / 128, (HEADS * HID) / 64);
    run_cvt(gat2_Wl, nullptr, Wbf, (long long)HID * HEADS * HID, stream);
    wmma_gemm_kernel<<<g, 128, 0, stream>>>(Abf, Wbf, gat2_bl, nullptr, nullptr,
                                            gl, MG, HEADS * HID, HID, 0);
    run_cvt(gat2_Wr, nullptr, Wbf, (long long)HID * HEADS * HID, stream);
    wmma_gemm_kernel<<<g, 128, 0, stream>>>(Abf, Wbf, gat2_br, nullptr, nullptr,
                                            gr, MG, HEADS * HID, HID, 0);
  }
  float* z2 = (float*)(ws + SZF + ((size_t)64 << 20));
  gatv2_att_kernel<<<BGRAPH, 256, 0, stream>>>(gl, gr, gat2_att, gat2_bias,
                                               bn4_g, bn4_b, z2);

  // ---- final FC ----
  fc_kernel<<<BGRAPH, 256, 0, stream>>>(z2, fc_W, fc_b, (float*)d_out);
}
